// MGRU_25426206392872
// MI455X (gfx1250) — matrix operand order repaired and checked by reading, not compiled, not run on hardware
//
#include <hip/hip_runtime.h>
#include <hip/hip_bf16.h>

// MGRU (minimal-GRU with 4-step delayed gate state) for MI455X / gfx1250.
//   1) mgru_proj : xz = x@Wz+bz, xh = x@Wh+bh   (bf16 WMMA, f32 out, bandwidth bound)
//   2) mgru_scan : sequential recurrence, 2 workgroups (16 batch rows each),
//                  Uz/Uh in VGPR fragments, h kept in registers (lane-stationary),
//                  delay ring + z*h staging in padded LDS (b128 frag loads),
//                  2x [16,256]x[256,256] bf16 WMMA GEMMs per timestep
//   3) mgru_head : y = hs@Wo+bo  (hs stored bf16 by the scan)
//
// Workspace (d_ws): xz f32 [B*L*H] | xh f32 [B*L*H] | hs bf16 [B*L*H] = 320MB.

typedef __attribute__((ext_vector_type(16))) __bf16 v16bf;
typedef __attribute__((ext_vector_type(8)))  __bf16 v8bf;
typedef __attribute__((ext_vector_type(8)))  float  v8f;

#define B_  32
#define L_  4096
#define I_  128
#define H_  256
#define O_  128
#define DELAY_ 4            // delay line length is DELAY_+1 = 5
#define ST_ (H_ + 8)        // padded LDS row stride in bf16: 264 (=132 dwords,
                            // 132 % 64 == 4 -> 4-dword bank rotation per row,
                            // 528B row keeps 16B alignment for b128 loads)

__device__ __forceinline__ v8f wmma_bf16(v16bf a, v16bf b, v8f c) {
  // D = A(16x32 bf16) * B(32x16 bf16) + C(16x16 f32)
  return __builtin_amdgcn_wmma_f32_16x16x32_bf16(
      /*neg_a=*/false, a, /*neg_b=*/false, b,
      /*c_mod=*/(short)0, c, /*reuse_a=*/false, /*reuse_b=*/false);
}

// A-matrix (16-bit, 16x32) fragment from a row-major row pointer.
// Per ISA 7.12.2 the per-lane data is two contiguous 16B chunks:
//   chunk0 = row[hi + 0..7], chunk1 = row[hi + 16..23], hi = 8*(lane>=16).
__device__ __forceinline__ v16bf load_a_frag(const __bf16* rowk /*row + k*32 + hi*/) {
  v8bf lo = *(const v8bf*)(rowk);
  v8bf hh = *(const v8bf*)(rowk + 16);
  return __builtin_shufflevector(lo, hh, 0, 1, 2, 3, 4, 5, 6, 7,
                                         8, 9, 10, 11, 12, 13, 14, 15);
}

// ---------------------------------------------------------------------------
// Kernel 1: input projections. Block = 256 thr (8 waves). Each wave computes a
// 16x16 tile of BOTH xz and xh (A-fragments reused). Grid: (B*L/128, H/16).
// ---------------------------------------------------------------------------
__global__ __launch_bounds__(256) void mgru_proj(
    const float* __restrict__ x,
    const float* __restrict__ Wz, const float* __restrict__ bz,
    const float* __restrict__ Wh, const float* __restrict__ bh,
    float* __restrict__ xz, float* __restrict__ xh)
{
  const int lane = threadIdx.x & 31;
  const int wave = threadIdx.x >> 5;
  const int row0 = blockIdx.x * 128 + wave * 16;  // rows of [B*L]
  const int n0   = blockIdx.y * 16;               // cols of H
  const int m    = lane & 15;
  const int hi8  = (lane >> 4) << 3;

  // A fragments: x[row0+m, k]  (K = I_ = 128 -> 4 k-tiles), f32->bf16 inline
  v16bf a[4];
#pragma unroll
  for (int k = 0; k < 4; ++k) {
    const float* xr = x + (size_t)(row0 + m) * I_ + k * 32 + hi8;
#pragma unroll
    for (int r = 0; r < 4; ++r) {           // chunk0: K = hi8 + 2r, 2r+1
      a[k][2 * r]     = (__bf16)xr[2 * r];
      a[k][2 * r + 1] = (__bf16)xr[2 * r + 1];
    }
#pragma unroll
    for (int r = 4; r < 8; ++r) {           // chunk1: K = hi8 + 16 + ...
      a[k][2 * r]     = (__bf16)xr[16 + 2 * (r - 4)];
      a[k][2 * r + 1] = (__bf16)xr[16 + 2 * (r - 4) + 1];
    }
  }

  v8f accz = {0.f, 0.f, 0.f, 0.f, 0.f, 0.f, 0.f, 0.f};
  v8f acch = {0.f, 0.f, 0.f, 0.f, 0.f, 0.f, 0.f, 0.f};
#pragma unroll
  for (int k = 0; k < 4; ++k) {
    // B fragment: N = lane&15, element e -> K = hi8 + e + (e&8)
    const float* wz = Wz + (size_t)(k * 32 + hi8) * H_ + n0 + (lane & 15);
    const float* wh = Wh + (size_t)(k * 32 + hi8) * H_ + n0 + (lane & 15);
    v16bf bzf, bhf;
#pragma unroll
    for (int v = 0; v < 8; ++v) {
      bzf[2 * v] = (__bf16)wz[(2 * v + ((2 * v) & 8)) * H_]; bzf[2 * v + 1] = (__bf16)wz[(2 * v + 1 + ((2 * v) & 8)) * H_];
      bhf[2 * v] = (__bf16)wh[(2 * v + ((2 * v) & 8)) * H_]; bhf[2 * v + 1] = (__bf16)wh[(2 * v + 1 + ((2 * v) & 8)) * H_];
    }
    accz = wmma_bf16(a[k], bzf, accz);
    acch = wmma_bf16(a[k], bhf, acch);
  }

  const int nl = lane & 15;
  const float bzv = bz[n0 + nl];
  const float bhv = bh[n0 + nl];
#pragma unroll
  for (int r = 0; r < 8; ++r) {
    int row = row0 + r + hi8;                  // C layout: M = r + 8*hi
    size_t o = (size_t)row * H_ + n0 + nl;     // N = lane&15
    xz[o] = accz[r] + bzv;
    xh[o] = acch[r] + bhv;
  }
}

// ---------------------------------------------------------------------------
// Kernel 2: sequential scan. Grid = 2 blocks (16 batch rows each), 512 thr =
// 16 waves; wave w owns hidden columns [16w, 16w+16). Uz/Uh live in VGPRs.
// h_{t-1} is lane-stationary -> kept in registers. LDS: padded bf16 delay
// ring (A for the z gate) and padded bf16 z*h staging (A for the candidate).
// ---------------------------------------------------------------------------
__global__ __launch_bounds__(512) void mgru_scan(
    const float* __restrict__ xz, const float* __restrict__ xh,
    const float* __restrict__ Uz, const float* __restrict__ Uh,
    __bf16* __restrict__ hs)
{
  __shared__ __bf16 s_hist[(DELAY_ + 1) * 16 * ST_];  // ~41 KB delay ring
  __shared__ __bf16 s_zh[16 * ST_];                   //  ~8 KB z*h staging

  const int lane = threadIdx.x & 31;
  const int wave = threadIdx.x >> 5;      // 0..15 -> N tile
  const int n0   = wave * 16;
  const int b0   = blockIdx.x * 16;       // batch-row group
  const int nl   = lane & 15;
  const int hi8  = (lane >> 4) << 3;      // +8 row offset for upper half-wave
  const int m    = lane & 15;

  // zero-init delay ring (reference: hist = 0); zh is written before read
  for (int i = threadIdx.x; i < (DELAY_ + 1) * 16 * ST_; i += blockDim.x)
    s_hist[i] = (__bf16)0.f;

  // Register-resident weight fragments: K = H_ = 256 -> 8 k-tiles per matrix.
  v16bf uzf[8], uhf[8];
#pragma unroll
  for (int k = 0; k < 8; ++k) {
    const float* uz = Uz + (size_t)(k * 32 + hi8) * H_ + n0 + nl;
    const float* uh = Uh + (size_t)(k * 32 + hi8) * H_ + n0 + nl;
#pragma unroll
    for (int v = 0; v < 8; ++v) {
      uzf[k][2 * v] = (__bf16)uz[(2 * v + ((2 * v) & 8)) * H_]; uzf[k][2 * v + 1] = (__bf16)uz[(2 * v + 1 + ((2 * v) & 8)) * H_];
      uhf[k][2 * v] = (__bf16)uh[(2 * v + ((2 * v) & 8)) * H_]; uhf[k][2 * v + 1] = (__bf16)uh[(2 * v + 1 + ((2 * v) & 8)) * H_];
    }
  }

  // lane-stationary h_{t-1}: this lane computed exactly these (b,n) elements
  float hp[8];
#pragma unroll
  for (int r = 0; r < 8; ++r) hp[r] = 0.f;

  __syncthreads();  // hist zero-init visible

  for (int t = 0; t < L_; ++t) {
    const int s = t % (DELAY_ + 1);       // ring slot: read h_{t-5}, write h_t

    // Stream this step's pre-activations; prefetch next step's rows.
    float xzt[8], xht[8];
#pragma unroll
    for (int r = 0; r < 8; ++r) {
      int b = b0 + r + hi8;
      size_t o = ((size_t)b * L_ + t) * H_ + n0 + nl;
      xzt[r] = xz[o];
      xht[r] = xh[o];
      if (t + 1 < L_) {
        __builtin_prefetch(&xz[o + H_], 0, 0);   // -> global_prefetch_b8
        __builtin_prefetch(&xh[o + H_], 0, 0);
      }
    }

    // ---- z = sigmoid(xz_t + h_{t-5} @ Uz) ----
    v8f acc = {0.f, 0.f, 0.f, 0.f, 0.f, 0.f, 0.f, 0.f};
    const __bf16* hd = &s_hist[s * 16 * ST_] + m * ST_ + hi8;
#pragma unroll
    for (int k = 0; k < 8; ++k)
      acc = wmma_bf16(load_a_frag(hd + k * 32), uzf[k], acc);

    float zv[8];
#pragma unroll
    for (int r = 0; r < 8; ++r) {
      int bl = r + hi8;
      float pre = acc[r] + xzt[r];
      zv[r] = 1.f / (1.f + __expf(-pre));
      s_zh[bl * ST_ + n0 + nl] = (__bf16)(zv[r] * hp[r]);
    }
    __syncthreads();  // orders hist reads above vs. writes below; zh visible

    // ---- h_tilde = tanh(xh_t + (z*h) @ Uh) ----
    v8f acc2 = {0.f, 0.f, 0.f, 0.f, 0.f, 0.f, 0.f, 0.f};
    const __bf16* zh = &s_zh[0] + m * ST_ + hi8;
#pragma unroll
    for (int k = 0; k < 8; ++k)
      acc2 = wmma_bf16(load_a_frag(zh + k * 32), uhf[k], acc2);

#pragma unroll
    for (int r = 0; r < 8; ++r) {
      int bl = r + hi8;
      float ht = tanhf(acc2[r] + xht[r]);
      float hn = (1.f - zv[r]) * hp[r] + zv[r] * ht;
      hp[r] = hn;                                            // h_t stays in regs
      s_hist[s * 16 * ST_ + bl * ST_ + n0 + nl] = (__bf16)hn; // h_t -> ring
      int b = b0 + bl;
      hs[((size_t)b * L_ + t) * H_ + n0 + nl] = (__bf16)hn;  // feed head GEMM
    }
    __syncthreads();  // hist writes visible before next step's reads;
                      // zh reads done before next step's zh writes
  }
}

// ---------------------------------------------------------------------------
// Kernel 3: head GEMM  y[b,l,:] = hs[b,l,:] @ Wo + bo.
// Grid: (B*L/128, O/16), block 256 (8 waves, one 16x16 tile each).
// ---------------------------------------------------------------------------
__global__ __launch_bounds__(256) void mgru_head(
    const __bf16* __restrict__ hs,
    const float* __restrict__ Wo, const float* __restrict__ bo,
    float* __restrict__ y)
{
  const int lane = threadIdx.x & 31;
  const int wave = threadIdx.x >> 5;
  const int row0 = blockIdx.x * 128 + wave * 16;
  const int n0   = blockIdx.y * 16;
  const int m    = lane & 15;
  const int hi8  = (lane >> 4) << 3;

  v8f acc = {0.f, 0.f, 0.f, 0.f, 0.f, 0.f, 0.f, 0.f};
  const __bf16* hr0 = hs + (size_t)(row0 + m) * H_ + hi8;
#pragma unroll
  for (int k = 0; k < 8; ++k) {             // K = H_ = 256
    v16bf a = load_a_frag(hr0 + k * 32);    // two 16B global chunks
    const float* wo = Wo + (size_t)(k * 32 + hi8) * O_ + n0 + (lane & 15);
    v16bf bf;
#pragma unroll
    for (int v = 0; v < 8; ++v) {
      bf[2 * v] = (__bf16)wo[(2 * v + ((2 * v) & 8)) * O_]; bf[2 * v + 1] = (__bf16)wo[(2 * v + 1 + ((2 * v) & 8)) * O_];
    }
    acc = wmma_bf16(a, bf, acc);
  }

  const int nl = lane & 15;
  const float bov = bo[n0 + nl];
#pragma unroll
  for (int r = 0; r < 8; ++r) {
    int row = row0 + r + hi8;
    y[(size_t)row * O_ + n0 + nl] = acc[r] + bov;
  }
}

// ---------------------------------------------------------------------------
extern "C" void kernel_launch(void* const* d_in, const int* in_sizes, int n_in,
                              void* d_out, int out_size, void* d_ws, size_t ws_size,
                              hipStream_t stream) {
  (void)in_sizes; (void)n_in; (void)out_size; (void)ws_size;
  const float* x  = (const float*)d_in[0];
  const float* Wz = (const float*)d_in[1];
  const float* Uz = (const float*)d_in[2];
  const float* bz = (const float*)d_in[3];
  const float* Wh = (const float*)d_in[4];
  const float* Uh = (const float*)d_in[5];
  const float* bh = (const float*)d_in[6];
  const float* Wo = (const float*)d_in[7];
  const float* bo = (const float*)d_in[8];
  float* y = (float*)d_out;

  const size_t BLH = (size_t)B_ * L_ * H_;
  float*  xz = (float*)d_ws;
  float*  xh = xz + BLH;
  __bf16* hs = (__bf16*)(xh + BLH);

  dim3 projGrid((B_ * L_) / 128, H_ / 16);          // 1024 x 16
  mgru_proj<<<projGrid, 256, 0, stream>>>(x, Wz, bz, Wh, bh, xz, xh);

  mgru_scan<<<dim3(B_ / 16), 512, 0, stream>>>(xz, xh, Uz, Uh, hs);

  dim3 headGrid((B_ * L_) / 128, O_ / 16);          // 1024 x 8
  mgru_head<<<headGrid, 256, 0, stream>>>(hs, Wo, bo, y);
}
